// Net_51067161149794
// MI455X (gfx1250) — matrix operand order repaired and checked by reading, not compiled, not run on hardware
//
#include <hip/hip_runtime.h>
#include <hip/hip_bf16.h>
#include <cstdint>
#include <cstddef>

// ---------------------------------------------------------------------------
// SplineCNN forward for MI455X (gfx1250, wave32, WMMA).
//
//   * basis kernel: per-edge degree-1 B-spline weights b[E][8] + kernel ids
//     wi[E][8] (u8), plus dst-degree counts.
//   * CSR by dst (count -> block scan -> bucket scatter).
//   * conv layer = ONE fused kernel per layer:
//       - workgroup = 16 dst nodes (N = 20000 = 1250 * 16 tiles exactly)
//       - Z tile [16][125*FCHUNK] accumulated in LDS with ds_add_f32 atomics,
//         one edge per thread (tile's CSR range is contiguous)
//       - contraction vs W via v_wmma_f32_16x16x32_f16 (f16 in, f32 accum),
//         8 waves split (n-tile x k-section), LDS reduction of partial C
//       - fused epilogue: *rdeg + x@root + bias + ELU (fast __expf)
//   * weights pre-swizzled to exact WMMA B-fragment order (128-bit loads).
//   * linear layers: same WMMA path; 512->2 head + log_softmax scalar.
// ---------------------------------------------------------------------------

typedef __attribute__((ext_vector_type(16))) _Float16 v16h;
typedef __attribute__((ext_vector_type(8)))  float    v8f;

constexpr int NN   = 20000;   // nodes
constexpr int EE   = 320000;  // edges
constexpr int KKER = 125;     // 5^3 kernel matrices
constexpr int MT   = NN / 16; // 1250 node tiles (exact)

// ---------------------------------------------------------------------------
// Phase 0: spline basis + degree counts
// ---------------------------------------------------------------------------
__global__ void basis_deg_kernel(const float* __restrict__ pseudo,
                                 const int* __restrict__ dstArr,
                                 float* __restrict__ bw,
                                 unsigned char* __restrict__ wt,
                                 int* __restrict__ cnt) {
  int e = blockIdx.x * blockDim.x + threadIdx.x;
  if (e >= EE) return;
  float fr[3]; int lo[3];
#pragma unroll
  for (int d = 0; d < 3; ++d) {
    float p = pseudo[e * 3 + d];
    p = fminf(fmaxf(p, 0.f), 1.f) * 4.f;        // * (KS-1)
    float fl = fminf(floorf(p), 3.f);            // clip to KS-2
    lo[d] = (int)fl;
    fr[d] = p - fl;
  }
#pragma unroll
  for (int t = 0; t < 8; ++t) {
    int b0 = t & 1, b1 = (t >> 1) & 1, b2 = (t >> 2) & 1;
    float w = (b0 ? fr[0] : (1.f - fr[0])) *
              (b1 ? fr[1] : (1.f - fr[1])) *
              (b2 ? fr[2] : (1.f - fr[2]));
    int k = (lo[0] + b0) + (lo[1] + b1) * 5 + (lo[2] + b2) * 25;
    bw[e * 8 + t] = w;
    wt[e * 8 + t] = (unsigned char)k;
  }
  atomicAdd(&cnt[dstArr[e]], 1);
}

// rdeg = 1 / max(count, 1)   (epilogue uses a multiply, not an IEEE divide)
__global__ void deg_kernel(const int* __restrict__ cnt, float* __restrict__ rdeg) {
  int n = blockIdx.x * blockDim.x + threadIdx.x;
  if (n < NN) rdeg[n] = 1.f / fmaxf((float)cnt[n], 1.f);
}

// single-block scan over NN counts -> CSR offsets
__global__ void scan_kernel(const int* __restrict__ cnt, int* __restrict__ off) {
  __shared__ int part[256];
  __shared__ int pfx[257];
  int t = threadIdx.x;
  const int per = (NN + 255) / 256;
  int lo = t * per, hi = min(lo + per, NN);
  int s = 0;
  for (int i = lo; i < hi; ++i) s += cnt[i];
  part[t] = s;
  __syncthreads();
  if (t == 0) {
    int acc = 0;
    for (int i = 0; i < 256; ++i) { pfx[i] = acc; acc += part[i]; }
    pfx[256] = acc;
  }
  __syncthreads();
  int acc = pfx[t];
  for (int i = lo; i < hi; ++i) { off[i] = acc; acc += cnt[i]; }
  if (t == 0) off[NN] = pfx[256];
}

__global__ void edge_sort_kernel(const int* __restrict__ dstArr,
                                 const int* __restrict__ csr_off,
                                 int* __restrict__ cursor,
                                 int* __restrict__ esort) {
  int e = blockIdx.x * blockDim.x + threadIdx.x;
  if (e >= EE) return;
  int d = dstArr[e];
  int p = atomicAdd(&cursor[d], 1);
  esort[csr_off[d] + p] = e;
}

// ---------------------------------------------------------------------------
// Weight swizzles into WMMA B-fragment order:
//   frag element (lane, idx): K = kk*32 + (lane>>4)*16 + idx, O = nt*16 + (lane&15)
//   storage: [..., kk, nt][lane*16 + idx]  (16 contiguous halves per lane)
// ---------------------------------------------------------------------------
template <int FIN, int FOUT, int FCHUNK>
__global__ void convert_conv_w(const float* __restrict__ W, _Float16* __restrict__ Wh) {
  constexpr int KKD = KKER * FCHUNK;
  constexpr int KKP = ((KKD + 31) / 32) * 32;
  constexpr int NCHUNK = FIN / FCHUNK;
  constexpr int KSTEPS = KKP / 32;
  constexpr int NT = FOUT / 16;
  const int total = NCHUNK * KSTEPS * NT * 512;
  for (int i = blockIdx.x * blockDim.x + threadIdx.x; i < total;
       i += gridDim.x * blockDim.x) {
    int idx  = i & 15;
    int lane = (i >> 4) & 31;
    int frag = i >> 9;                 // (ch, kk, nt)
    int nt = frag % NT;
    int kk = (frag / NT) % KSTEPS;
    int ch = frag / (NT * KSTEPS);
    int krow = kk * 32 + (lane >> 4) * 16 + idx;
    int o    = nt * 16 + (lane & 15);
    float v = 0.f;
    if (krow < KKD) {
      int k = krow / FCHUNK, fl = krow % FCHUNK;
      v = W[(k * FIN + ch * FCHUNK + fl) * FOUT + o];
    }
    Wh[i] = (_Float16)v;
  }
}

template <int FIN, int FOUT>
__global__ void convert_lin_w(const float* __restrict__ W, _Float16* __restrict__ Wh) {
  constexpr int KSTEPS = FIN / 32;
  constexpr int NT = FOUT / 16;
  const int total = KSTEPS * NT * 512;
  for (int i = blockIdx.x * blockDim.x + threadIdx.x; i < total;
       i += gridDim.x * blockDim.x) {
    int idx  = i & 15;
    int lane = (i >> 4) & 31;
    int frag = i >> 9;
    int nt = frag % NT;
    int kk = frag / NT;
    int krow = kk * 32 + (lane >> 4) * 16 + idx;
    int o    = nt * 16 + (lane & 15);
    Wh[i] = (_Float16)W[krow * FOUT + o];
  }
}

// ---------------------------------------------------------------------------
// Fused spline conv layer
// ---------------------------------------------------------------------------
template <int FIN, int FOUT, int FCHUNK>
__global__ __launch_bounds__(256) void spline_conv_kernel(
    const float* __restrict__ hin, const int* __restrict__ srcArr,
    const int* __restrict__ dstArr,
    const int* __restrict__ csr_off, const int* __restrict__ esort,
    const float* __restrict__ bw, const unsigned char* __restrict__ wt,
    const float* __restrict__ rdeg, const _Float16* __restrict__ Wh,
    const float* __restrict__ root, const float* __restrict__ bias,
    float* __restrict__ hout) {
  constexpr int KKD = KKER * FCHUNK;
  constexpr int KKP = ((KKD + 31) / 32) * 32;   // pad K dim to 32
  constexpr int NCHUNK = FIN / FCHUNK;
  constexpr int KSTEPS = KKP / 32;
  constexpr int NT = FOUT / 16;                 // n-tiles (2 or 4)
  constexpr int KSPL = 8 / NT;                  // waves per n-tile

  __shared__ float Zt[16 * KKP];                // Z tile; reused for C-reduce

  const int tile = blockIdx.x;
  const int lane = threadIdx.x & 31;
  const int wave = threadIdx.x >> 5;
  const int nt   = wave % NT;
  const int ksec = wave / NT;
  const int am   = lane & 15;
  const int ag   = lane >> 4;

  // the tile's 16 CSR buckets are contiguous in esort
  const int e0 = csr_off[tile * 16];
  const int ne = csr_off[tile * 16 + 16] - e0;

  v8f cacc = {};

  for (int ch = 0; ch < NCHUNK; ++ch) {
    // ---- zero Z tile (covers the K padding too) ----
    for (int i = threadIdx.x; i < 16 * KKP; i += 256) Zt[i] = 0.f;
    __syncthreads();

    // ---- edge scatter into LDS (ds_add_f32), one edge per thread ----
    for (int el = threadIdx.x; el < ne; el += 256) {
      int e = esort[e0 + el];
      int nl = dstArr[e] - tile * 16;           // local dst row
      int s  = srcArr[e];
      float xv[FCHUNK];
#pragma unroll
      for (int f = 0; f < FCHUNK; ++f) xv[f] = hin[s * FIN + ch * FCHUNK + f];
      float* zrow = &Zt[nl * KKP];
#pragma unroll
      for (int t = 0; t < 8; ++t) {
        float w = bw[e * 8 + t];
        int k = wt[e * 8 + t];
#pragma unroll
        for (int f = 0; f < FCHUNK; ++f)
          atomicAdd(&zrow[k * FCHUNK + f], w * xv[f]);
      }
    }
    __syncthreads();

    // ---- WMMA contraction: wave (nt, ksec) handles its k-step slice ----
    const int kb = (KSTEPS * ksec) / KSPL;
    const int ke = (KSTEPS * (ksec + 1)) / KSPL;
    for (int kk = kb; kk < ke; ++kk) {
      const int kbase = kk * 32;
      v16h a;
#pragma unroll
      for (int v = 0; v < 8; ++v)
#pragma unroll
        for (int h2 = 0; h2 < 2; ++h2) {
          int idx = v * 2 + h2;
          int ka = ag * 16 + idx;
          a[idx] = (_Float16)Zt[am * KKP + kbase + ka];
        }
      // B fragment: pre-swizzled, 16 contiguous halves per lane (2x b128)
      const v16h b = *(const v16h*)&Wh[((((size_t)ch * KSTEPS + kk) * NT) + nt) * 512 +
                                       lane * 16];
      cacc = __builtin_amdgcn_wmma_f32_16x16x32_f16(false, a, false, b,
                                                    (short)0, cacc, false, false);
    }
    __syncthreads();
  }

  // ---- reduce partial C across k-sections (reuse Zt; needs 8KB <= Zt size) ----
  if (KSPL > 1) {
    float* red = Zt;
#pragma unroll
    for (int v = 0; v < 8; ++v) red[(wave * 32 + lane) * 8 + v] = cacc[v];
    __syncthreads();
    if (ksec == 0) {
#pragma unroll
      for (int v = 0; v < 8; ++v) {
        float s = cacc[v];
        for (int q = 1; q < KSPL; ++q)
          s += red[(((q * NT + nt) * 32) + lane) * 8 + v];
        cacc[v] = s;
      }
    }
  }

  // ---- fused epilogue: *rdeg + x@root + bias + ELU ----
  if (ksec == 0) {
    const int o = nt * 16 + am;
#pragma unroll
    for (int v = 0; v < 8; ++v) {
      int m = v + ag * 8;
      int node = tile * 16 + m;
      float val = cacc[v] * rdeg[node];
      float rt = 0.f;
      for (int f = 0; f < FIN; ++f) rt = fmaf(hin[node * FIN + f], root[f * FOUT + o], rt);
      val += rt + bias[o];
      val = val > 0.f ? val : (__expf(val) - 1.f);   // ELU (hw v_exp_f32)
      hout[node * FOUT + o] = val;
    }
  }
}

// ---------------------------------------------------------------------------
// WMMA linear layer (+ ELU)
// ---------------------------------------------------------------------------
template <int FIN, int FOUT, bool DOELU>
__global__ __launch_bounds__(256) void linear_kernel(
    const float* __restrict__ hin, const _Float16* __restrict__ Wh,
    const float* __restrict__ bias, float* __restrict__ hout) {
  constexpr int KSTEPS = FIN / 32;
  constexpr int NT = FOUT / 16;
  const int lane = threadIdx.x & 31;
  const int wave = threadIdx.x >> 5;
  const int am = lane & 15, ag = lane >> 4;
  const int total = MT * NT;
  const int stride = gridDim.x * 8;
  for (int item = blockIdx.x * 8 + wave; item < total; item += stride) {
    int mt = item / NT, nt = item % NT;
    v8f c = {};
    for (int kk = 0; kk < KSTEPS; ++kk) {
      const int kbase = kk * 32;
      v16h a;
#pragma unroll
      for (int v = 0; v < 8; ++v)
#pragma unroll
        for (int h2 = 0; h2 < 2; ++h2) {
          int idx = v * 2 + h2;
          int ka = ag * 16 + idx;
          a[idx] = (_Float16)hin[(mt * 16 + am) * FIN + kbase + ka];
        }
      const v16h b = *(const v16h*)&Wh[(((size_t)kk * NT) + nt) * 512 + lane * 16];
      c = __builtin_amdgcn_wmma_f32_16x16x32_f16(false, a, false, b,
                                                 (short)0, c, false, false);
    }
    const int o = nt * 16 + am;
#pragma unroll
    for (int v = 0; v < 8; ++v) {
      int node = mt * 16 + v + ag * 8;
      float val = c[v] + bias[o];
      if (DOELU) val = val > 0.f ? val : (__expf(val) - 1.f);
      hout[node * FOUT + o] = val;
    }
  }
}

// ---------------------------------------------------------------------------
// 512 -> 2 head + log_softmax
// ---------------------------------------------------------------------------
__global__ void head_kernel(const float* __restrict__ h, const float* __restrict__ W,
                            const float* __restrict__ b, float* __restrict__ out) {
  int n = blockIdx.x * blockDim.x + threadIdx.x;
  if (n >= NN) return;
  float l0 = b[0], l1 = b[1];
  for (int f = 0; f < 512; ++f) {
    float hv = h[n * 512 + f];
    l0 = fmaf(hv, W[f * 2 + 0], l0);
    l1 = fmaf(hv, W[f * 2 + 1], l1);
  }
  float mx = fmaxf(l0, l1);
  float lse = mx + __logf(__expf(l0 - mx) + __expf(l1 - mx));
  out[n * 2 + 0] = l0 - lse;
  out[n * 2 + 1] = l1 - lse;
}

// ---------------------------------------------------------------------------
// Host orchestration
// Input order assumption (setup_inputs insertion order, tuples flattened):
//   0:x 1:edge_index 2:pseudo
//   3..5:c11(W,root,b) 6..8:c12 9..10:l13 11..13:c21 14..16:c22 17..18:l23
//   19..21:c31 22..24:c32 25..26:l33 27..29:c41 30..32:c42 33..34:l43
//   35..36:l44(W,b)
// ---------------------------------------------------------------------------
extern "C" void kernel_launch(void* const* d_in, const int* in_sizes, int n_in,
                              void* d_out, int out_size, void* d_ws, size_t ws_size,
                              hipStream_t stream) {
  (void)in_sizes; (void)n_in; (void)out_size; (void)ws_size;
  const float* x      = (const float*)d_in[0];
  const int*   ei     = (const int*)d_in[1];
  const float* pseudo = (const float*)d_in[2];
  const int* srcArr = ei;
  const int* dstArr = ei + EE;

  auto cW = [&](int i) { return (const float*)d_in[i]; };

  // workspace carve-out (256B aligned); total ~98 MB
  char* ws = (char*)d_ws;
  size_t off = 0;
  auto take = [&](size_t bytes) {
    void* p = ws + off;
    off = (off + bytes + 255) & ~(size_t)255;
    return p;
  };
  int*   cnt    = (int*)take((size_t)NN * 4);
  int*   cursor = (int*)take((size_t)NN * 4);
  int*   csr    = (int*)take((size_t)(NN + 1) * 4);
  float* rdeg   = (float*)take((size_t)NN * 4);
  int*   esort  = (int*)take((size_t)EE * 4);
  float* bwgt   = (float*)take((size_t)EE * 8 * 4);
  unsigned char* wtap = (unsigned char*)take((size_t)EE * 8);
  float* hA = (float*)take((size_t)NN * 512 * 4);
  float* hB = (float*)take((size_t)NN * 512 * 4);
  _Float16* Wh = (_Float16*)take((size_t)1 << 20);   // 1MB, max layer needs 512KB

  hipMemsetAsync(cnt, 0, (size_t)NN * 4, stream);
  hipMemsetAsync(cursor, 0, (size_t)NN * 4, stream);

  basis_deg_kernel<<<(EE + 255) / 256, 256, 0, stream>>>(pseudo, dstArr, bwgt, wtap, cnt);
  scan_kernel<<<1, 256, 0, stream>>>(cnt, csr);
  deg_kernel<<<(NN + 255) / 256, 256, 0, stream>>>(cnt, rdeg);
  edge_sort_kernel<<<(EE + 255) / 256, 256, 0, stream>>>(dstArr, csr, cursor, esort);

  auto conv = [&](auto finT, auto foutT, auto fchT, const float* hin, int wi,
                  float* hout) {
    constexpr int FIN = decltype(finT)::value;
    constexpr int FOUT = decltype(foutT)::value;
    constexpr int FCH = decltype(fchT)::value;
    convert_conv_w<FIN, FOUT, FCH><<<512, 256, 0, stream>>>(cW(wi), Wh);
    spline_conv_kernel<FIN, FOUT, FCH><<<MT, 256, 0, stream>>>(
        hin, srcArr, dstArr, csr, esort, bwgt, wtap, rdeg, Wh,
        cW(wi + 1), cW(wi + 2), hout);
  };
  auto lin = [&](auto finT, auto foutT, const float* hin, int wi, float* hout) {
    constexpr int FIN = decltype(finT)::value;
    constexpr int FOUT = decltype(foutT)::value;
    convert_lin_w<FIN, FOUT><<<128, 256, 0, stream>>>(cW(wi), Wh);
    constexpr int NT = FOUT / 16;
    int blocks = (MT * NT + 7) / 8;
    linear_kernel<FIN, FOUT, true><<<blocks, 256, 0, stream>>>(hin, Wh, cW(wi + 1), hout);
  };

  using i1  = std::integral_constant<int, 1>;
  using i8  = std::integral_constant<int, 8>;
  using i16 = std::integral_constant<int, 16>;
  using i32 = std::integral_constant<int, 32>;
  using i64 = std::integral_constant<int, 64>;
  using i512 = std::integral_constant<int, 512>;

  conv(i1{},  i32{}, i1{}, x,  3,  hA);   // c11: 1 -> 32
  conv(i32{}, i64{}, i8{}, hA, 6,  hB);   // c12: 32 -> 64
  lin (i64{}, i16{},       hB, 9,  hA);   // l13: 64 -> 16
  conv(i16{}, i32{}, i8{}, hA, 11, hB);   // c21
  conv(i32{}, i64{}, i8{}, hB, 14, hA);   // c22
  lin (i64{}, i16{},       hA, 17, hB);   // l23
  conv(i16{}, i32{}, i8{}, hB, 19, hA);   // c31
  conv(i32{}, i64{}, i8{}, hA, 22, hB);   // c32
  lin (i64{}, i16{},       hB, 25, hA);   // l33
  conv(i16{}, i32{}, i8{}, hA, 27, hB);   // c41
  conv(i32{}, i64{}, i8{}, hB, 30, hA);   // c42
  lin (i64{}, i512{},      hA, 33, hB);   // l43: 64 -> 512
  head_kernel<<<(NN + 255) / 256, 256, 0, stream>>>(hB, cW(35), cW(36), (float*)d_out);
}
